// PropagatePerceptualSim_23845658427938
// MI455X (gfx1250) — compile-verified, matrix operand order checked by reading, not run on hardware
//
#include <hip/hip_runtime.h>

typedef __attribute__((ext_vector_type(16))) _Float16 v16h;
typedef __attribute__((ext_vector_type(8)))  float    v8f;
typedef __attribute__((ext_vector_type(4)))  unsigned v4u;

#define B_   8
#define H_   512
#define W_   512
#define HW_  (H_ * W_)
#define NPIX (B_ * HW_)

__device__ __forceinline__ int imax_(int a, int b) { return a > b ? a : b; }
__device__ __forceinline__ int imin_(int a, int b) { return a < b ? a : b; }

// ---------------------------------------------------------------------------
// Repack w2 (OIHW f32 [64][64][3][3]) -> f16 [co][tap=ky*3+kx][ci], so each
// WMMA B-fragment is a pair of contiguous 16B loads.
// ---------------------------------------------------------------------------
__global__ void k_packw2(const float* __restrict__ w2, _Float16* __restrict__ w2h) {
    int i = blockIdx.x * 256 + threadIdx.x;            // over 64*64*9 = 36864
    if (i >= 64 * 64 * 9) return;
    int kx = i % 3;
    int ky = (i / 3) % 3;
    int ci = (i / 9) % 64;
    int co = i / (9 * 64);
    w2h[(co * 9 + ky * 3 + kx) * 64 + ci] = (_Float16)w2[i];
}

// ---------------------------------------------------------------------------
// conv1_1: normalize (ImageNet mean/std) + 3->64 3x3 conv + bias + ReLU.
// Output: NHWC f16 (ci contiguous) so conv2 A-fragments are K-contiguous.
// One thread = one pixel, all 64 output channels.
// ---------------------------------------------------------------------------
__global__ void __launch_bounds__(256) k_conv1(const float* __restrict__ img,
                                               const float* __restrict__ w1,
                                               const float* __restrict__ b1,
                                               _Float16* __restrict__ feat1) {
    __shared__ float sw[64 * 27 + 64];
    // stage weights permuted to [co][tap][c]
    for (int i = threadIdx.x; i < 64 * 27; i += 256) {
        int co = i / 27, r = i % 27;     // w1 OIHW: i = co*27 + c*9 + ky*3 + kx
        int c = r / 9, t = r % 9;
        sw[co * 27 + t * 3 + c] = w1[i];
    }
    for (int i = threadIdx.x; i < 64; i += 256) sw[64 * 27 + i] = b1[i];
    __syncthreads();

    int pix = blockIdx.x * 256 + threadIdx.x;
    int x = pix % W_, y = (pix / W_) % H_, b = pix / HW_;

    const float mean[3]   = {0.485f, 0.456f, 0.406f};
    const float invstd[3] = {1.0f / 0.229f, 1.0f / 0.224f, 1.0f / 0.225f};
    float xv[27];
#pragma unroll
    for (int ky = 0; ky < 3; ++ky) {
        int yy = y + ky - 1;
#pragma unroll
        for (int kx = 0; kx < 3; ++kx) {
            int xx = x + kx - 1;
            bool inb = (yy >= 0) && (yy < H_) && (xx >= 0) && (xx < W_);
#pragma unroll
            for (int c = 0; c < 3; ++c) {
                float v = 0.0f;
                if (inb) v = (img[((size_t)(b * 3 + c) * H_ + yy) * W_ + xx] - mean[c]) * invstd[c];
                xv[(ky * 3 + kx) * 3 + c] = v;
            }
        }
    }

    unsigned* outp = (unsigned*)(feat1 + (size_t)pix * 64);
    for (int p = 0; p < 32; ++p) {                      // pairs of output channels
        float a0 = sw[64 * 27 + 2 * p];
        float a1 = sw[64 * 27 + 2 * p + 1];
        const float* wr0 = &sw[(2 * p) * 27];
        const float* wr1 = &sw[(2 * p + 1) * 27];
#pragma unroll
        for (int t = 0; t < 27; ++t) {
            a0 = fmaf(wr0[t], xv[t], a0);
            a1 = fmaf(wr1[t], xv[t], a1);
        }
        a0 = fmaxf(a0, 0.0f);
        a1 = fmaxf(a1, 0.0f);
        union { _Float16 h[2]; unsigned u; } pk;
        pk.h[0] = (_Float16)a0;
        pk.h[1] = (_Float16)a1;
        outp[p] = pk.u;
    }
}

// ---------------------------------------------------------------------------
// conv1_2: 64->64 3x3 conv as implicit GEMM with V_WMMA_F32_16X16X32_F16.
// Block = 8 waves = 32 pixels (x-contiguous) x 64 output channels.
// Wave (w): pixel tile (w&1)*16, channel tile (w>>1)*16.
// A (16x32 f16): feature tile from LDS.  B (32x16 f16): repacked weights.
// A-layout per ISA: lane 0-15 -> M=lane, K 0-7 & 16-23; lane 16-31 -> K 8-15 & 24-31
//   => two ds_load_b128 at byte offsets (kb*2) and (kb*2 + 32). Same for B with N.
// ---------------------------------------------------------------------------
__global__ void __launch_bounds__(256) k_conv2(const _Float16* __restrict__ feat1,
                                               const _Float16* __restrict__ w2h,
                                               const float* __restrict__ b2,
                                               _Float16* __restrict__ feat2) {
    __shared__ __align__(16) _Float16 sf[3 * 34 * 64];   // [row 0..2][x -1..32][ci], 13056 B

    int p0 = blockIdx.x * 32;
    int x0 = p0 % W_, y = (p0 / W_) % H_, b = p0 / HW_;

    // Stage 3 rows x 34 cols x 64ch of f16 features (as dwords) with zero padding.
    const unsigned* g32 = (const unsigned*)feat1;
    unsigned* s32 = (unsigned*)sf;
    for (int e = threadIdx.x; e < 3 * 34 * 32; e += 256) {
        int ci2 = e % 32;
        int xx = (e / 32) % 34;
        int r = e / (32 * 34);
        int gy = y + r - 1, gx = x0 + xx - 1;
        unsigned v = 0u;
        if (gy >= 0 && gy < H_ && gx >= 0 && gx < W_)
            v = g32[(size_t)((b * H_ + gy) * W_ + gx) * 32 + ci2];
        s32[(r * 34 + xx) * 32 + ci2] = v;
    }
    __syncthreads();

    int lane  = threadIdx.x & 31;
    int wv    = threadIdx.x >> 5;
    int mtile = (wv & 1) * 16;        // pixel sub-tile
    int ntile = (wv >> 1) * 16;       // channel sub-tile
    int l15   = lane & 15;            // M row (A) / N col (B)
    int kb    = (lane >> 4) * 8;      // K sub-block select per half-wave

    v8f acc = {};
#pragma unroll
    for (int t = 0; t < 9; ++t) {
        int dy = t / 3, dx = t % 3;
#pragma unroll
        for (int cb = 0; cb < 64; cb += 32) {
            // A fragment: feature row (dy), pixel (mtile + l15 + dx), K = cb+kb ...
            const v4u* ap = (const v4u*)&sf[((dy * 34) + (mtile + l15 + dx)) * 64 + cb + kb];
            union { v4u q[2]; v16h h; } ua;
            ua.q[0] = ap[0];
            ua.q[1] = ap[2];          // +32 bytes: K 16..23 / 24..31 half
            // B fragment: weights [co = ntile+l15][tap t][ci = cb+kb ...]
            const v4u* bp = (const v4u*)&w2h[((size_t)(ntile + l15) * 9 + t) * 64 + cb + kb];
            union { v4u q[2]; v16h h; } ub;
            ub.q[0] = bp[0];
            ub.q[1] = bp[2];
            acc = __builtin_amdgcn_wmma_f32_16x16x32_f16(
                false, ua.h, false, ub.h, (short)0, acc, false, false);
        }
    }

    // D layout: lane 0-15 -> N=lane, M=v; lane 16-31 -> N=lane-16, M=v+8.
    int n  = lane & 15;
    int co = ntile + n;
    float bias = b2[co];
#pragma unroll
    for (int vi = 0; vi < 8; ++vi) {
        int m  = vi + 8 * (lane >> 4);
        int px = x0 + mtile + m;
        float val = fmaxf(acc[vi] + bias, 0.0f);
        feat2[(size_t)((b * H_ + y) * W_ + px) * 64 + co] = (_Float16)val;
    }
}

// ---------------------------------------------------------------------------
// Vertical / horizontal feature similarity: exp(-||df||^2 / 500).
// ---------------------------------------------------------------------------
__global__ void __launch_bounds__(256) k_sims(const _Float16* __restrict__ feat2,
                                              float* __restrict__ vsim,
                                              float* __restrict__ hsim) {
    int pix = blockIdx.x * 256 + threadIdx.x;
    int x = pix % W_, y = (pix / W_) % H_;
    const v4u* f0 = (const v4u*)(feat2 + (size_t)pix * 64);
    v4u c[8];
#pragma unroll
    for (int i = 0; i < 8; ++i) c[i] = f0[i];

    if (y < H_ - 1) {
        const v4u* fd = (const v4u*)(feat2 + (size_t)(pix + W_) * 64);
        float s = 0.0f;
#pragma unroll
        for (int i = 0; i < 8; ++i) {
            union { v4u q; _Float16 h[8]; } ua, ub;
            ua.q = c[i];
            ub.q = fd[i];
#pragma unroll
            for (int j = 0; j < 8; ++j) {
                float df = (float)ua.h[j] - (float)ub.h[j];
                s = fmaf(df, df, s);
            }
        }
        vsim[pix] = __expf(-s * (1.0f / 500.0f));
    }
    if (x < W_ - 1) {
        const v4u* fr = (const v4u*)(feat2 + (size_t)(pix + 1) * 64);
        float s = 0.0f;
#pragma unroll
        for (int i = 0; i < 8; ++i) {
            union { v4u q; _Float16 h[8]; } ua, ub;
            ua.q = c[i];
            ub.q = fr[i];
#pragma unroll
            for (int j = 0; j < 8; ++j) {
                float df = (float)ua.h[j] - (float)ub.h[j];
                s = fmaf(df, df, s);
            }
        }
        hsim[pix] = __expf(-s * (1.0f / 500.0f));
    }
}

// ---------------------------------------------------------------------------
// Mask erode (3x3 ones conv, zero pad, >1) and knowness init: cur0 = 1 - eroded.
// (mask resize 512->512 bilinear is identity.)
// ---------------------------------------------------------------------------
__global__ void __launch_bounds__(256) k_init(const float* __restrict__ mask,
                                              float* __restrict__ cur) {
    int pix = blockIdx.x * 256 + threadIdx.x;
    int x = pix % W_, y = (pix / W_) % H_, b = pix / HW_;
    float s = 0.0f;
#pragma unroll
    for (int dy = -1; dy <= 1; ++dy) {
        int yy = y + dy;
#pragma unroll
        for (int dx = -1; dx <= 1; ++dx) {
            int xx = x + dx;
            if (yy >= 0 && yy < H_ && xx >= 0 && xx < W_)
                s += mask[((size_t)b * H_ + yy) * W_ + xx];
        }
    }
    cur[pix] = (s > 1.0f) ? 0.0f : 1.0f;
}

// ---------------------------------------------------------------------------
// One propagation iteration (replicate-padded directional max with similarity).
// ---------------------------------------------------------------------------
__global__ void __launch_bounds__(256) k_prop(const float* __restrict__ cin,
                                              const float* __restrict__ vsim,
                                              const float* __restrict__ hsim,
                                              float* __restrict__ cout) {
    int pix = blockIdx.x * 256 + threadIdx.x;
    int x = pix % W_, y = (pix / W_) % H_, b = pix / HW_;
    int rb = b * H_;

    int ym1 = imax_(y - 1, 0);
    int yp1 = imin_(y + 1, H_ - 1);
    int ys  = imin_(y, H_ - 2);
    int xm1 = imax_(x - 1, 0);
    int xp1 = imin_(x + 1, W_ - 1);
    int xs  = imin_(x, W_ - 2);

    float c    = cin[pix];
    float ntop = cin[(rb + ym1) * W_ + x]   * vsim[(rb + ym1) * W_ + x];
    float nbot = cin[(rb + yp1) * W_ + x]   * vsim[(rb + ys)  * W_ + x];
    float nlef = cin[(rb + y)   * W_ + xm1] * hsim[(rb + y)   * W_ + xm1];
    float nrig = cin[(rb + y)   * W_ + xp1] * hsim[(rb + y)   * W_ + xs];

    cout[pix] = fmaxf(c, fmaxf(fmaxf(ntop, nbot), fmaxf(nlef, nrig)));
}

// ---------------------------------------------------------------------------
// out = min(mask, 1 - cur)
// ---------------------------------------------------------------------------
__global__ void __launch_bounds__(256) k_final(const float* __restrict__ mask,
                                               const float* __restrict__ cur,
                                               float* __restrict__ out) {
    int pix = blockIdx.x * 256 + threadIdx.x;
    out[pix] = fminf(mask[pix], 1.0f - cur[pix]);
}

// ---------------------------------------------------------------------------
extern "C" void kernel_launch(void* const* d_in, const int* in_sizes, int n_in,
                              void* d_out, int out_size, void* d_ws, size_t ws_size,
                              hipStream_t stream) {
    const float* real_img = (const float*)d_in[0];
    // d_in[1] = pred_img: unused by the reference forward pass
    const float* mask = (const float*)d_in[2];
    const float* w1 = (const float*)d_in[3];
    const float* b1 = (const float*)d_in[4];
    const float* w2 = (const float*)d_in[5];
    const float* b2 = (const float*)d_in[6];
    // d_in[7] = erode_w: all-ones, folded into k_init

    char* ws = (char*)d_ws;
    _Float16* feat1 = (_Float16*)(ws + 0ull);                 // 256 MB  NHWC f16
    _Float16* feat2 = (_Float16*)(ws + 268435456ull);         // 256 MB  NHWC f16
    _Float16* w2h   = (_Float16*)(ws + 536870912ull);         // 72 KB   repacked weights
    float*    vsim  = (float*)   (ws + 537001984ull);         // 8 MB
    float*    hsim  = (float*)   (ws + 545390592ull);         // 8 MB
    float*    cur0  = (float*)   (ws + 553779200ull);         // 8 MB
    float*    cur1  = (float*)   (ws + 562167808ull);         // 8 MB
    float* out = (float*)d_out;

    k_packw2<<<(64 * 64 * 9 + 255) / 256, 256, 0, stream>>>(w2, w2h);
    k_conv1<<<NPIX / 256, 256, 0, stream>>>(real_img, w1, b1, feat1);
    k_conv2<<<NPIX / 32, 256, 0, stream>>>(feat1, w2h, b2, feat2);
    k_sims<<<NPIX / 256, 256, 0, stream>>>(feat2, vsim, hsim);
    k_init<<<NPIX / 256, 256, 0, stream>>>(mask, cur0);
    for (int it = 0; it < 10; ++it) {
        const float* ci = (it & 1) ? cur1 : cur0;
        float*       co = (it & 1) ? cur0 : cur1;
        k_prop<<<NPIX / 256, 256, 0, stream>>>(ci, vsim, hsim, co);
    }
    // 10 iterations (even) -> final result lands back in cur0
    k_final<<<NPIX / 256, 256, 0, stream>>>(mask, cur0, out);
}
